// MultiHeadAttention_6296422056702
// MI455X (gfx1250) — matrix operand order repaired and checked by reading, not compiled, not run on hardware
//
#include <hip/hip_runtime.h>
#include <hip/hip_bf16.h>

// ---------------------------------------------------------------------------
// MHA for B=2, T=2048, C=1024, H=16, D=64 on gfx1250 (CDNA5, wave32, WMMA).
// All matmuls: v_wmma_f32_16x16x32_bf16.  Operand staging: triple-buffered
// global_load_async_to_lds_b128 (ASYNCcnt) + ds_load_b128 fragment reads,
// one s_barrier per K-step.
// ---------------------------------------------------------------------------

typedef __bf16 bf16;
typedef __attribute__((ext_vector_type(16))) __bf16 v16bf;
typedef __attribute__((ext_vector_type(8)))  __bf16 v8bf;
typedef __attribute__((ext_vector_type(4)))  __bf16 v4bf;
typedef __attribute__((ext_vector_type(8)))  float  v8f;

#define TTOT 4096   // B*T rows
#define CDIM 1024   // embed dim
#define TB   2048   // tokens per batch
#define NH   16
#define HD   64

static __device__ __forceinline__ v8f wmma_bf16f32(v16bf a, v16bf b, v8f c) {
  return __builtin_amdgcn_wmma_f32_16x16x32_bf16(false, a, false, b, (short)0, c,
                                                 false, false);
}

// Async copy of one 16B chunk global -> LDS (ASYNCcnt-tracked DMA).
static __device__ __forceinline__ void async_b128(unsigned lds_addr,
                                                  const void* gaddr) {
  asm volatile("global_load_async_to_lds_b128 %0, %1, off"
               :: "v"(lds_addr), "v"(gaddr) : "memory");
}
#define WAIT_ASYNC(n) asm volatile("s_wait_asynccnt " #n ::: "memory")

static __device__ __forceinline__ unsigned lds_off(const void* p) {
  return (unsigned)(uintptr_t)p;  // LDS aperture: low 32 bits = LDS byte addr
}

// A-operand (16x32): lane l16 = row; lanes 0-15 take K {0..7,16..23},
// lanes 16-31 take K {8..15,24..31}.  base = &src[row*pitch + k0].
static __device__ __forceinline__ v16bf load_afrag(const bf16* base, int hf) {
  union { v16bf v; v8bf h[2]; } u;
  u.h[0] = *(const v8bf*)(base + hf * 8);
  u.h[1] = *(const v8bf*)(base + 16 + hf * 8);
  return u.v;
}

// B-operand (32x16) from B^T rows: lanes 0-15 take K {0..7,16..23}, lanes 16-31
// take K {8..15,24..31}.  base = &bt[n*pitch + k0].
static __device__ __forceinline__ v16bf load_bfrag(const bf16* base, int hf) {
  union { v16bf v; v8bf h[2]; } u;
  u.h[0] = *(const v8bf*)(base + hf * 8);
  u.h[1] = *(const v8bf*)(base + 16 + hf * 8);
  return u.v;
}

// ---------------------------------------------------------------------------
__global__ __launch_bounds__(256) void cvt_f32_bf16(const float* __restrict__ in,
                                                    bf16* __restrict__ out, int n4) {
  int i = blockIdx.x * blockDim.x + threadIdx.x;
  if (i < n4) {
    float4 f = ((const float4*)in)[i];
    v4bf o = { (bf16)f.x, (bf16)f.y, (bf16)f.z, (bf16)f.w };
    ((v4bf*)out)[i] = o;
  }
}

// ---------------------------------------------------------------------------
// C[m,n] = sum_k A[m,k] * Bt[n,k].  A: [4096,1024], Bt: [1024,1024] row-major.
// MODE 0: f32 -> outF[m*1024+n]
// MODE 1: bf16 (scaled) -> outB[m*1024+n]
// MODE 2: bf16 transposed per-head -> outB[(b*1024+n)*2048 + (m%2048)]
// Block: 8 waves; block tile 128(M) x 128(N); wave tile 64x32; K-step 32;
// triple-buffered async LDS staging (pitch 40 elems vs bank conflicts).
template <int MODE>
__global__ __launch_bounds__(256) void gemm_bf16(const bf16* __restrict__ A,
                                                 const bf16* __restrict__ Bt,
                                                 float* __restrict__ outF,
                                                 bf16* __restrict__ outB,
                                                 float scale) {
  __shared__ bf16 sA[3][128 * 40];
  __shared__ bf16 sB[3][128 * 40];

  const int tid = threadIdx.x;
  const int lane = tid & 31, w = tid >> 5;
  const int hf = lane >> 4, l16 = lane & 15;
  const int m0b = blockIdx.y * 128;
  const int n0b = blockIdx.x * 128;
  const int wm = (w >> 2) * 64;   // wave M offset in block tile
  const int wn = (w & 3) * 32;    // wave N offset in block tile

  // staging: 128 rows x 4 chunks(16B) per tile = 512 chunks -> 2 per thread
  const int r4 = tid >> 2, c4 = tid & 3;
  const size_t gA0 = (size_t)(m0b + r4) * CDIM + c4 * 8;
  const size_t gA1 = (size_t)(m0b + r4 + 64) * CDIM + c4 * 8;
  const size_t gB0 = (size_t)(n0b + r4) * CDIM + c4 * 8;
  const size_t gB1 = (size_t)(n0b + r4 + 64) * CDIM + c4 * 8;
  const unsigned l0 = r4 * 80 + c4 * 16;          // bytes, pitch 40 elems
  const unsigned l1 = (r4 + 64) * 80 + c4 * 16;

  // prologue stage: buffer 0, k0 = 0
  async_b128(lds_off(sA[0]) + l0, A + gA0);
  async_b128(lds_off(sA[0]) + l1, A + gA1);
  async_b128(lds_off(sB[0]) + l0, Bt + gB0);
  async_b128(lds_off(sB[0]) + l1, Bt + gB1);

  v8f acc[4][2] = {};
  for (int it = 0; it < CDIM / 32; ++it) {
    const int buf = it % 3;
    if (it + 1 < CDIM / 32) {
      const int nb = (it + 1) % 3;
      const int k1 = (it + 1) * 32;
      async_b128(lds_off(sA[nb]) + l0, A + gA0 + k1);
      async_b128(lds_off(sA[nb]) + l1, A + gA1 + k1);
      async_b128(lds_off(sB[nb]) + l0, Bt + gB0 + k1);
      async_b128(lds_off(sB[nb]) + l1, Bt + gB1 + k1);
      WAIT_ASYNC(0x4);
    } else {
      WAIT_ASYNC(0x0);
    }
    __syncthreads();

    v16bf a[4], bb[2];
#pragma unroll
    for (int mi = 0; mi < 4; ++mi)
      a[mi] = load_afrag(&sA[buf][(wm + mi * 16 + l16) * 40], hf);
#pragma unroll
    for (int nj = 0; nj < 2; ++nj)
      bb[nj] = load_bfrag(&sB[buf][(wn + nj * 16 + l16) * 40], hf);
#pragma unroll
    for (int mi = 0; mi < 4; ++mi)
#pragma unroll
      for (int nj = 0; nj < 2; ++nj)
        acc[mi][nj] = wmma_bf16f32(a[mi], bb[nj], acc[mi][nj]);
  }

#pragma unroll
  for (int mi = 0; mi < 4; ++mi)
#pragma unroll
    for (int nj = 0; nj < 2; ++nj)
#pragma unroll
      for (int r = 0; r < 8; ++r) {
        int gr = m0b + wm + mi * 16 + hf * 8 + r;
        int gc = n0b + wn + nj * 16 + l16;
        float v = acc[mi][nj][r];
        if (MODE == 0) {
          outF[(size_t)gr * CDIM + gc] = v;
        } else if (MODE == 1) {
          outB[(size_t)gr * CDIM + gc] = (bf16)(v * scale);
        } else {
          int bb2 = gr >> 11, t = gr & (TB - 1);
          outB[((size_t)bb2 * CDIM + gc) * TB + t] = (bf16)v;
        }
      }
}

// ---------------------------------------------------------------------------
// Flash-style attention. Q,K: bf16 [4096,1024] (Q pre-scaled by D^-0.5).
// Vt: bf16 [2, 1024, 2048] = [b, h*64+d, t].  O: bf16 [4096,1024].
// Block = 8 waves, each wave owns 32 queries; all 8 waves share one (b,h), so
// K/V tiles are staged once per block via async-to-LDS (triple buffered,
// single barrier per s-step).
__global__ __launch_bounds__(256) void attn_kernel(const bf16* __restrict__ Q,
                                                   const bf16* __restrict__ K,
                                                   const bf16* __restrict__ Vt,
                                                   bf16* __restrict__ O) {
  __shared__ bf16 sK[3][32 * 72];  // [s_local][d], pitch 72
  __shared__ bf16 sV[3][64 * 40];  // [d][s_local], pitch 40
  __shared__ bf16 sP[8][32 * 40];  // per-wave P staging, pitch 40

  const int tid = threadIdx.x;
  const int lane = tid & 31, w = tid >> 5;
  const int hf = lane >> 4, l16 = lane & 15;
  const int bh = blockIdx.x >> 3;
  const int b = bh >> 4, h = bh & 15;
  const int q0 = (blockIdx.x & 7) * 256 + w * 32;

  const bf16* Qb = Q + (size_t)b * TB * CDIM + (size_t)h * HD;
  const bf16* Kb = K + (size_t)b * TB * CDIM + (size_t)h * HD;
  const bf16* Vb = Vt + ((size_t)b * CDIM + h * HD) * TB;

  // staging coords: K tile 32x64 (8 chunks/row), V tile 64x32 (4 chunks/row)
  const int kr = tid >> 3, kc = tid & 7;
  const int vr = tid >> 2, vc = tid & 3;
  const size_t gK = (size_t)kr * CDIM + kc * 8;
  const unsigned lK = kr * 144 + kc * 16;
  const size_t gV = (size_t)vr * TB + vc * 8;
  const unsigned lV = vr * 80 + vc * 16;

  // Q fragments (loaded once, direct from global)
  v16bf qa[2][2];
#pragma unroll
  for (int mi = 0; mi < 2; ++mi)
#pragma unroll
    for (int kd = 0; kd < 2; ++kd)
      qa[mi][kd] = load_afrag(Qb + (size_t)(q0 + mi * 16 + l16) * CDIM + kd * 32, hf);

  // prologue stage for s0 = 0
  async_b128(lds_off(sK[0]) + lK, Kb + gK);
  async_b128(lds_off(sV[0]) + lV, Vb + gV);

  v8f accO[2][4] = {};
  float rm[2][8], rl[2][8];
#pragma unroll
  for (int mi = 0; mi < 2; ++mi)
#pragma unroll
    for (int r = 0; r < 8; ++r) { rm[mi][r] = -1e30f; rl[mi][r] = 0.f; }

  for (int it = 0; it < TB / 32; ++it) {
    const int buf = it % 3;
    const int s0 = it * 32;
    if (it + 1 < TB / 32) {
      const int nb = (it + 1) % 3;
      async_b128(lds_off(sK[nb]) + lK, Kb + gK + (size_t)(s0 + 32) * CDIM);
      async_b128(lds_off(sV[nb]) + lV, Vb + gV + (s0 + 32));
      WAIT_ASYNC(0x2);
    } else {
      WAIT_ASYNC(0x0);
    }
    __syncthreads();

    // ---- S = Q @ K^T for a 32x32 tile (K frags from LDS) ----
    v16bf kb[2][2];
#pragma unroll
    for (int sj = 0; sj < 2; ++sj)
#pragma unroll
      for (int kd = 0; kd < 2; ++kd)
        kb[sj][kd] = load_bfrag(&sK[buf][(sj * 16 + l16) * 72 + kd * 32], hf);

    v8f S[2][2] = {};
#pragma unroll
    for (int mi = 0; mi < 2; ++mi)
#pragma unroll
      for (int sj = 0; sj < 2; ++sj)
#pragma unroll
        for (int kd = 0; kd < 2; ++kd)
          S[mi][sj] = wmma_bf16f32(qa[mi][kd], kb[sj][kd], S[mi][sj]);

    // ---- online softmax (rows live in 16-lane halves) ----
#pragma unroll
    for (int mi = 0; mi < 2; ++mi) {
      float corr[8];
#pragma unroll
      for (int r = 0; r < 8; ++r) {
        float v = fmaxf(S[mi][0][r], S[mi][1][r]);
#pragma unroll
        for (int msk = 1; msk < 16; msk <<= 1) v = fmaxf(v, __shfl_xor(v, msk, 32));
        float nm = fmaxf(rm[mi][r], v);
        corr[r] = __expf(rm[mi][r] - nm);
        rm[mi][r] = nm;
      }
#pragma unroll
      for (int sj = 0; sj < 2; ++sj)
#pragma unroll
        for (int r = 0; r < 8; ++r)
          S[mi][sj][r] = __expf(S[mi][sj][r] - rm[mi][r]);
#pragma unroll
      for (int r = 0; r < 8; ++r) {
        float s = S[mi][0][r] + S[mi][1][r];
#pragma unroll
        for (int msk = 1; msk < 16; msk <<= 1) s += __shfl_xor(s, msk, 32);
        rl[mi][r] = rl[mi][r] * corr[r] + s;
      }
#pragma unroll
      for (int dj = 0; dj < 4; ++dj)
#pragma unroll
        for (int r = 0; r < 8; ++r)
          accO[mi][dj][r] *= corr[r];
    }

    // ---- P (C-layout f32) -> per-wave LDS bf16 -> A-fragments ----
    __builtin_amdgcn_wave_barrier();
#pragma unroll
    for (int mi = 0; mi < 2; ++mi)
#pragma unroll
      for (int sj = 0; sj < 2; ++sj)
#pragma unroll
        for (int r = 0; r < 8; ++r)
          sP[w][(mi * 16 + hf * 8 + r) * 40 + sj * 16 + l16] = (bf16)S[mi][sj][r];
    __builtin_amdgcn_wave_barrier();

    v16bf pa[2];
#pragma unroll
    for (int mi = 0; mi < 2; ++mi)
      pa[mi] = load_afrag(&sP[w][(mi * 16 + l16) * 40], hf);
    __builtin_amdgcn_wave_barrier();

    // ---- O += P @ V  (V frags from LDS) ----
    v16bf vb[4];
#pragma unroll
    for (int dj = 0; dj < 4; ++dj)
      vb[dj] = load_bfrag(&sV[buf][(dj * 16 + l16) * 40], hf);
#pragma unroll
    for (int mi = 0; mi < 2; ++mi)
#pragma unroll
      for (int dj = 0; dj < 4; ++dj)
        accO[mi][dj] = wmma_bf16f32(pa[mi], vb[dj], accO[mi][dj]);
  }

  // ---- normalize and store O (bf16, [B,T,H*D]) ----
#pragma unroll
  for (int mi = 0; mi < 2; ++mi)
#pragma unroll
    for (int r = 0; r < 8; ++r) {
      float inv = 1.0f / rl[mi][r];
      int gr = b * TB + q0 + mi * 16 + hf * 8 + r;
#pragma unroll
      for (int dj = 0; dj < 4; ++dj)
        O[(size_t)gr * CDIM + h * HD + dj * 16 + l16] =
            (bf16)(accO[mi][dj][r] * inv);
    }
}

// ---------------------------------------------------------------------------
extern "C" void kernel_launch(void* const* d_in, const int* in_sizes, int n_in,
                              void* d_out, int out_size, void* d_ws, size_t ws_size,
                              hipStream_t stream) {
  const float* x  = (const float*)d_in[0];
  const float* wq = (const float*)d_in[1];
  const float* wk = (const float*)d_in[2];
  const float* wv = (const float*)d_in[3];
  const float* wo = (const float*)d_in[4];

  char* ws = (char*)d_ws;
  size_t off = 0;
  auto alloc = [&](size_t bytes) -> void* {
    void* p = ws + off;
    off += (bytes + 255) & ~(size_t)255;
    return p;
  };
  bf16* xb  = (bf16*)alloc((size_t)TTOT * CDIM * 2);
  bf16* wqb = (bf16*)alloc((size_t)CDIM * CDIM * 2);
  bf16* wkb = (bf16*)alloc((size_t)CDIM * CDIM * 2);
  bf16* wvb = (bf16*)alloc((size_t)CDIM * CDIM * 2);
  bf16* wob = (bf16*)alloc((size_t)CDIM * CDIM * 2);
  bf16* qb  = (bf16*)alloc((size_t)TTOT * CDIM * 2);
  bf16* kbuf= (bf16*)alloc((size_t)TTOT * CDIM * 2);
  bf16* vtb = (bf16*)alloc((size_t)TTOT * CDIM * 2);
  bf16* ob  = (bf16*)alloc((size_t)TTOT * CDIM * 2);

  // f32 -> bf16 conversions
  cvt_f32_bf16<<<(TTOT * CDIM / 4) / 256, 256, 0, stream>>>(x, xb, TTOT * CDIM / 4);
  cvt_f32_bf16<<<(CDIM * CDIM / 4) / 256, 256, 0, stream>>>(wq, wqb, CDIM * CDIM / 4);
  cvt_f32_bf16<<<(CDIM * CDIM / 4) / 256, 256, 0, stream>>>(wk, wkb, CDIM * CDIM / 4);
  cvt_f32_bf16<<<(CDIM * CDIM / 4) / 256, 256, 0, stream>>>(wv, wvb, CDIM * CDIM / 4);
  cvt_f32_bf16<<<(CDIM * CDIM / 4) / 256, 256, 0, stream>>>(wo, wob, CDIM * CDIM / 4);

  dim3 gg(CDIM / 128, TTOT / 128);
  // Q = x@wq^T (scaled by D^-0.5), K = x@wk^T, V = x@wv^T (stored transposed)
  gemm_bf16<1><<<gg, 256, 0, stream>>>(xb, wqb, nullptr, qb, 0.125f);
  gemm_bf16<1><<<gg, 256, 0, stream>>>(xb, wkb, nullptr, kbuf, 1.0f);
  gemm_bf16<2><<<gg, 256, 0, stream>>>(xb, wvb, nullptr, vtb, 1.0f);

  // attention: 32 (b,h) pairs x 8 query blocks
  attn_kernel<<<256, 256, 0, stream>>>(qb, kbuf, vtb, ob);

  // final projection to f32 output
  gemm_bf16<0><<<gg, 256, 0, stream>>>(ob, wob, (float*)d_out, nullptr, 1.0f);
}
